// Attention_8761733284386
// MI455X (gfx1250) — compile-verified, matrix operand order checked by reading, not run on hardware
//
#include <hip/hip_runtime.h>
#include <hip/hip_bf16.h>

#define Bc   2
#define Tc   2048
#define DIMc 2048
#define Hc   32
#define KVHc 8
#define HDc  64
#define QPKV (Hc / KVHc)

typedef __attribute__((ext_vector_type(16))) __bf16 v16bf;
typedef __attribute__((ext_vector_type(8)))  float  v8f;
typedef __attribute__((ext_vector_type(4)))  unsigned int v4u;
typedef __attribute__((ext_vector_type(4)))  int    v4i;
typedef __attribute__((ext_vector_type(8)))  int    v8i;

#if defined(__HIP_DEVICE_COMPILE__) && __has_builtin(__builtin_amdgcn_tensor_load_to_lds)
#define HAS_TDM 1
#else
#define HAS_TDM 0
#endif

__device__ __forceinline__ v8f wmma_bf16(v16bf a, v16bf b, v8f c) {
  // 8 args: (neg_a, A, neg_b, B, c_mod, C, reuse_a, reuse_b)
  return __builtin_amdgcn_wmma_f32_16x16x32_bf16(false, a, false, b, (short)0, c, false, false);
}

// ---------------------------------------------------------------------------
// TDM: async-DMA a 2D bf16 tile (tile_w x tile_h) from row-major global memory
// (row stride `row_stride` elements) into LDS at byte offset `lds_off`.
// D# packed per ISA 8.3/8.4. Padding: rows of 64 bf16 = 32 dwords, then pad
// 4 dwords (8 elements) -> LDS row stride of 72 elements, matching GAS below.
// Tracked by TENSORcnt.
// ---------------------------------------------------------------------------
#if HAS_TDM
__device__ __forceinline__ void tdm_load_tile_bf16(unsigned lds_off, const __bf16* gp,
                                                   unsigned tile_w, unsigned tile_h,
                                                   unsigned row_stride, unsigned tensor_rows) {
  unsigned long long ga = (unsigned long long)(const void*)gp;
  v4u g0;
  g0[0] = 1u;                                          // count=1, user descriptor
  g0[1] = lds_off;                                     // LDS byte address
  g0[2] = (unsigned)ga;                                // global_addr[31:0]
  g0[3] = (unsigned)((ga >> 32) & 0x01FFFFFFull) | (2u << 30); // addr[56:32] | type=2
  v8i g1;
  // data_size=2B (code 1), pad_enable, pad_interval=32dw (code 4), pad_amount=4dw (code 3)
  g1[0] = (1 << 16) | (1 << 20) | (4 << 22) | (3 << 25);
  g1[1] = (int)((row_stride & 0xFFFFu) << 16);                         // tensor_dim0[15:0]
  g1[2] = (int)(((row_stride >> 16) & 0xFFFFu) | ((tensor_rows & 0xFFFFu) << 16));
  g1[3] = (int)(((tensor_rows >> 16) & 0xFFFFu) | ((tile_w & 0xFFFFu) << 16));
  g1[4] = (int)(tile_h & 0xFFFFu);                                     // tile_dim1, tile_dim2=0
  g1[5] = (int)row_stride;                                             // tensor_dim0_stride[31:0]
  g1[6] = 0;                                                           // stride hi / dim1_stride lo
  g1[7] = 0;
  v4i gz = {0, 0, 0, 0};
#if defined(__clang_major__) && (__clang_major__ >= 23)
  v8i gz8 = {0, 0, 0, 0, 0, 0, 0, 0};
  __builtin_amdgcn_tensor_load_to_lds(g0, g1, gz, gz, gz8, 0);
#else
  __builtin_amdgcn_tensor_load_to_lds(g0, g1, gz, gz, 0);
#endif
}
template <int N>
__device__ __forceinline__ void tdm_wait() { __builtin_amdgcn_s_wait_tensorcnt(N); }
#else
__device__ __forceinline__ void tdm_load_tile_bf16(unsigned, const __bf16*, unsigned,
                                                   unsigned, unsigned, unsigned) {}
template <int N>
__device__ __forceinline__ void tdm_wait() {}
#endif

// A-style fragment: 16 rows x 32 k, tile stored row-major (stride ld) in LDS.
// ISA 7.12.2: lane&15 selects row; lane bit4 selects k-offset 0/8; VGPR j<4 holds
// k = koff+2j..+1, VGPR j>=4 holds k = 16+koff+2(j-4)..+1.
__device__ __forceinline__ v16bf frag_rowmaj(const __bf16* s, int row0, int ld, int k0, int lane) {
  int r = row0 + (lane & 15);
  int koff = (lane & 16) ? 8 : 0;
  const __bf16* base = s + r * ld + k0 + koff;
  v16bf f;
#pragma unroll
  for (int j = 0; j < 4; ++j) {
    f[2 * j]         = base[2 * j];
    f[2 * j + 1]     = base[2 * j + 1];
    f[8 + 2 * j]     = base[16 + 2 * j];
    f[8 + 2 * j + 1] = base[16 + 2 * j + 1];
  }
  return f;
}

// B-style fragment: 32 k x 16 cols, tile stored [k][n] row-major (stride ld) in LDS.
__device__ __forceinline__ v16bf frag_colmaj(const __bf16* s, int k0, int col0, int ld, int lane) {
  int c = col0 + (lane & 15);
  int koff = (lane & 16) ? 8 : 0;
  const __bf16* base = s + (k0 + koff) * ld + c;
  v16bf f;
#pragma unroll
  for (int j = 0; j < 4; ++j) {
    f[2 * j]         = base[(2 * j) * ld];
    f[2 * j + 1]     = base[(2 * j + 1) * ld];
    f[8 + 2 * j]     = base[(16 + 2 * j) * ld];
    f[8 + 2 * j + 1] = base[(16 + 2 * j + 1) * ld];
  }
  return f;
}

// ---------------------------------------------------------------------------
// Tiled WMMA GEMM: C[M,N] = A[M,K](bf16) @ B[K,N](bf16), f32 accumulate.
// Block tile 64x64, BK=64, 128 threads = 4 waves, each wave 32x32 (2x2 WMMA
// tiles). TDM double-buffered LDS staging: wave 0 DMAs tile i+1 while all
// waves consume tile i with WMMA.
// LDS layout (bytes): [0: As0][TILE_B: Bs0][2*TILE_B: As1][3*TILE_B: Bs1]
// ---------------------------------------------------------------------------
#define GAS 72                       // padded LDS row stride in elements
#define TILE_B (64 * GAS * 2)        // bytes per 64-row tile

template <bool STORE_BF16>
__global__ __launch_bounds__(128) void gemm_wmma(const __bf16* __restrict__ A,
                                                 const __bf16* __restrict__ Bm,
                                                 float* __restrict__ Cf,
                                                 __bf16* __restrict__ Cb,
                                                 int M, int N, int K) {
  extern __shared__ __align__(16) char smem[];   // 4 * TILE_B bytes

  int tid = threadIdx.x;
  int lane = tid & 31;
  int wave = tid >> 5;
  int m0 = blockIdx.y * 64;
  int n0 = blockIdx.x * 64;
  int wr = (wave >> 1) * 32;
  int wc = (wave & 1) * 32;
  v8f acc[2][2] = {};

#if HAS_TDM
  if (wave == 0) {   // prologue: DMA stage 0
    tdm_load_tile_bf16(0u, A + (size_t)m0 * K, 64, 64, (unsigned)K, (unsigned)(M - m0));
    tdm_load_tile_bf16((unsigned)TILE_B, Bm + n0, 64, 64, (unsigned)N, (unsigned)K);
  }
#endif

  int buf = 0;
  for (int kk = 0; kk < K; kk += 64) {
    // Current-stage buffer pointers (runtime select over integer offsets —
    // no constant LDS-pointer arrays, which the device linker rejects).
    __bf16* As = (__bf16*)(smem + (buf ? 2 * TILE_B : 0));
    __bf16* Bs = (__bf16*)(smem + (buf ? 3 * TILE_B : TILE_B));
#if HAS_TDM
    if (wave == 0) {
      if (kk + 64 < K) {   // DMA next stage into the other buffer, then gate on current
        unsigned aoff = buf ? 0u : 2u * TILE_B;
        unsigned boff = buf ? 1u * TILE_B : 3u * TILE_B;
        tdm_load_tile_bf16(aoff, A + (size_t)m0 * K + kk + 64, 64, 64,
                           (unsigned)K, (unsigned)(M - m0));
        tdm_load_tile_bf16(boff, Bm + (size_t)(kk + 64) * N + n0, 64, 64,
                           (unsigned)N, (unsigned)(K - kk - 64));
        tdm_wait<2>();
      } else {
        tdm_wait<0>();
      }
    }
#else
#pragma unroll
    for (int it = 0; it < 4; ++it) {
      int idx = tid * 8 + it * 1024;
      int r = idx >> 6, c = idx & 63;
      *(uint4*)&As[r * GAS + c] = *(const uint4*)(A + (size_t)(m0 + r) * K + kk + c);
      *(uint4*)&Bs[r * GAS + c] = *(const uint4*)(Bm + (size_t)(kk + r) * N + n0 + c);
    }
#endif
    __syncthreads();
#pragma unroll
    for (int k2 = 0; k2 < 64; k2 += 32) {
      v16bf a0 = frag_rowmaj(As, wr, GAS, k2, lane);
      v16bf a1 = frag_rowmaj(As, wr + 16, GAS, k2, lane);
      v16bf b0 = frag_colmaj(Bs, k2, wc, GAS, lane);
      v16bf b1 = frag_colmaj(Bs, k2, wc + 16, GAS, lane);
      acc[0][0] = wmma_bf16(a0, b0, acc[0][0]);
      acc[0][1] = wmma_bf16(a0, b1, acc[0][1]);
      acc[1][0] = wmma_bf16(a1, b0, acc[1][0]);
      acc[1][1] = wmma_bf16(a1, b1, acc[1][1]);
    }
    __syncthreads();   // reads done before this buffer is DMA-overwritten
    buf ^= 1;
  }

  // C/D layout: lane&15 = col; lane>>4 selects row block 0/8; VGPR r = row within block.
  int col = lane & 15;
  int rb = (lane >> 4) * 8;
#pragma unroll
  for (int ti = 0; ti < 2; ++ti)
#pragma unroll
    for (int tj = 0; tj < 2; ++tj)
#pragma unroll
      for (int r = 0; r < 8; ++r) {
        size_t gi = (size_t)(m0 + wr + ti * 16 + rb + r) * N + (n0 + wc + tj * 16 + col);
        float v = acc[ti][tj][r];
        if (STORE_BF16) Cb[gi] = (__bf16)v; else Cf[gi] = v;
      }
}

// ---------------------------------------------------------------------------
// Elementwise f32 -> bf16
// ---------------------------------------------------------------------------
__global__ void cvt_f32_bf16(const float* __restrict__ s, __bf16* __restrict__ d, int n) {
  int i = blockIdx.x * blockDim.x + threadIdx.x;
  if (i < n) d[i] = (__bf16)s[i];
}

// ---------------------------------------------------------------------------
// RoPE (optional) + transpose [b,t,h,hd] -> [b,h,t,hd], f32 math, bf16 out.
// ---------------------------------------------------------------------------
__global__ void rope_tr(const __bf16* __restrict__ src, const float* __restrict__ fc,
                        const float* __restrict__ fs, __bf16* __restrict__ dst,
                        int heads, int do_rope, float scale, int n) {
  int idx = blockIdx.x * blockDim.x + threadIdx.x;
  if (idx >= n) return;
  int d2 = idx & 31;          // HD/2 = 32
  int rem = idx >> 5;
  int hh = rem % heads;
  int rem2 = rem / heads;
  int t = rem2 % Tc;
  int b = rem2 / Tc;
  size_t si = ((size_t)(b * Tc + t) * heads + hh) * HDc + 2 * d2;
  float x0 = (float)src[si], x1 = (float)src[si + 1];
  float re = x0, ro = x1;
  if (do_rope) {
    float c = fc[t * 32 + d2], s = fs[t * 32 + d2];
    re = x0 * c - x1 * s;
    ro = x0 * s + x1 * c;
  }
  size_t di = ((size_t)(b * heads + hh) * Tc + t) * HDc + 2 * d2;
  dst[di]     = (__bf16)(re * scale);
  dst[di + 1] = (__bf16)(ro * scale);
}

// ---------------------------------------------------------------------------
// Causal flash attention. Grid: (T/64, B*H). Block: 128 threads = 4 waves.
// Each wave owns 16 query rows. K/V 32x64 tiles are TDM double-buffered; Q is
// TDM-staged once. Softmax stats live in LDS per row; scores round-trip
// through LDS to convert C-layout -> A-layout for the P@V WMMAs.
//
// Dynamic LDS layout (bytes):
//   0      Qs      64x72 bf16   (9216)
//   9216   Ks[0]   32x72 bf16   (4608)
//   13824  Vs[0]   32x72 bf16   (4608)
//   18432  Ks[1]   32x72 bf16   (4608)
//   23040  Vs[1]   32x72 bf16   (4608)
//   27648  Ss      4x16x32 f32  (8192)
//   35840  Ps      4x16x40 bf16 (5120)
//   40960  mrow/lrow/frow 3x64 f32 (768)   total 41728
// ---------------------------------------------------------------------------
#define FA_LDS_BYTES 41728

__global__ __launch_bounds__(128) void flash_attn(const __bf16* __restrict__ qT,
                                                  const __bf16* __restrict__ kT,
                                                  const __bf16* __restrict__ vT,
                                                  __bf16* __restrict__ out) {
  extern __shared__ __align__(16) char smem[];
  __bf16* Qs   = (__bf16*)smem;
  float*  Ss   = (float*)(smem + 27648);
  __bf16* Ps   = (__bf16*)(smem + 35840);
  float*  mrow = (float*)(smem + 40960);
  float*  lrow = mrow + 64;
  float*  frow = lrow + 64;

  int tid = threadIdx.x, lane = tid & 31, wave = tid >> 5;
  int qt = blockIdx.x, bh = blockIdx.y;
  int b = bh / Hc, h = bh % Hc, kvh = h / QPKV;
  int q0 = qt * 64;
  const __bf16* qbase = qT + ((size_t)(b * Hc + h) * Tc + q0) * HDc;
  const __bf16* kbase = kT + (size_t)(b * KVHc + kvh) * Tc * HDc;
  const __bf16* vbase = vT + (size_t)(b * KVHc + kvh) * Tc * HDc;

  // Stage Q (64x64) once.
#if HAS_TDM
  if (wave == 0) {
    tdm_load_tile_bf16(0u, qbase, 64, 64, 64, (unsigned)(Tc - q0));
    tdm_wait<0>();
  }
#else
#pragma unroll
  for (int it = 0; it < 4; ++it) {
    int idx = tid * 8 + it * 1024;
    int r = idx >> 6, c = idx & 63;
    *(uint4*)&Qs[r * 72 + c] = *(const uint4*)(qbase + (size_t)r * HDc + c);
  }
#endif
  if (tid < 64) { mrow[tid] = -3.0e38f; lrow[tid] = 0.f; }
  __syncthreads();

  v16bf aq0 = frag_rowmaj(Qs, wave * 16, 72, 0, lane);   // held across KV loop
  v16bf aq1 = frag_rowmaj(Qs, wave * 16, 72, 32, lane);
  v8f O[4] = {};
  int col = lane & 15, rb = (lane >> 4) * 8;
  int nkv = q0 + 64;  // causal: only KV cols < q0+64 are needed by this block

#if HAS_TDM
  if (wave == 0) {   // prologue: DMA KV stage 0
    tdm_load_tile_bf16(9216u, kbase, 64, 32, 64, (unsigned)Tc);
    tdm_load_tile_bf16(13824u, vbase, 64, 32, 64, (unsigned)Tc);
  }
#endif

  int buf = 0;
  for (int c0 = 0; c0 < nkv; c0 += 32) {
    // Current-stage buffer pointers (runtime select over integer offsets).
    __bf16* Ks = (__bf16*)(smem + (buf ? 18432 : 9216));
    __bf16* Vs = (__bf16*)(smem + (buf ? 23040 : 13824));
#if HAS_TDM
    if (wave == 0) {
      if (c0 + 32 < nkv) {
        unsigned koff = buf ? 9216u : 18432u;
        unsigned voff = buf ? 13824u : 23040u;
        tdm_load_tile_bf16(koff, kbase + (size_t)(c0 + 32) * HDc, 64, 32, 64,
                           (unsigned)(Tc - c0 - 32));
        tdm_load_tile_bf16(voff, vbase + (size_t)(c0 + 32) * HDc, 64, 32, 64,
                           (unsigned)(Tc - c0 - 32));
        tdm_wait<2>();
      } else {
        tdm_wait<0>();
      }
    }
#else
#pragma unroll
    for (int it = 0; it < 2; ++it) {
      int idx = tid * 8 + it * 1024;
      int r = idx >> 6, c = idx & 63;
      *(uint4*)&Ks[r * 72 + c] = *(const uint4*)(kbase + (size_t)(c0 + r) * HDc + c);
      *(uint4*)&Vs[r * 72 + c] = *(const uint4*)(vbase + (size_t)(c0 + r) * HDc + c);
    }
#endif
    __syncthreads();

    // S(16x32) = Q(16x64) @ K^T : B-frag "row" index is the KV position in Ks.
    v8f s0 = {}, s1 = {};
    {
      v16bf bk00 = frag_rowmaj(Ks, 0,  72, 0,  lane);
      v16bf bk01 = frag_rowmaj(Ks, 0,  72, 32, lane);
      v16bf bk10 = frag_rowmaj(Ks, 16, 72, 0,  lane);
      v16bf bk11 = frag_rowmaj(Ks, 16, 72, 32, lane);
      s0 = wmma_bf16(aq0, bk00, s0);
      s0 = wmma_bf16(aq1, bk01, s0);
      s1 = wmma_bf16(aq0, bk10, s1);
      s1 = wmma_bf16(aq1, bk11, s1);
    }
    float* Ssw = Ss + wave * 512;
#pragma unroll
    for (int r = 0; r < 8; ++r) {
      Ssw[(rb + r) * 32 + col]      = s0[r];
      Ssw[(rb + r) * 32 + 16 + col] = s1[r];
    }
    __syncthreads();

    // Online softmax: lanes 0..15 of each wave each own one query row.
    __bf16* Psw = Ps + wave * 640;
    if (lane < 16) {
      int row = lane;
      int qrow = q0 + wave * 16 + row;
      float m_old = mrow[wave * 16 + row];
      float mx = m_old;
#pragma unroll 4
      for (int j = 0; j < 32; ++j) {
        float s = Ssw[row * 32 + j];
        if (c0 + j > qrow) s = -3.0e38f;
        mx = fmaxf(mx, s);
      }
      float fac = __expf(m_old - mx);
      float sum = 0.f;
#pragma unroll 4
      for (int j = 0; j < 32; ++j) {
        float s = Ssw[row * 32 + j];
        float p = (c0 + j > qrow) ? 0.f : __expf(s - mx);
        sum += p;
        Psw[row * 40 + j] = (__bf16)p;
      }
      mrow[wave * 16 + row] = mx;
      lrow[wave * 16 + row] = lrow[wave * 16 + row] * fac + sum;
      frow[wave * 16 + row] = fac;
    }
    __syncthreads();

    // Rescale O by per-row factor, then O += P(16x32) @ V(32x64).
    float facv[8];
#pragma unroll
    for (int r = 0; r < 8; ++r) facv[r] = frow[wave * 16 + rb + r];
    v16bf pa = frag_rowmaj(Psw, 0, 40, 0, lane);
#pragma unroll
    for (int t4 = 0; t4 < 4; ++t4) {
#pragma unroll
      for (int r = 0; r < 8; ++r) O[t4][r] *= facv[r];
      v16bf bv = frag_colmaj(Vs, 0, t4 * 16, 72, lane);
      O[t4] = wmma_bf16(pa, bv, O[t4]);
    }
    __syncthreads();   // reads done before this KV buffer is DMA-overwritten
    buf ^= 1;
  }

  float linv[8];
#pragma unroll
  for (int r = 0; r < 8; ++r) linv[r] = 1.0f / lrow[wave * 16 + rb + r];
#pragma unroll
  for (int t4 = 0; t4 < 4; ++t4)
#pragma unroll
    for (int r = 0; r < 8; ++r) {
      int trow = q0 + wave * 16 + rb + r;
      out[((size_t)(b * Tc + trow)) * (Hc * HDc) + h * HDc + t4 * 16 + col] =
          (__bf16)(O[t4][r] * linv[r]);
    }
}

// ---------------------------------------------------------------------------
// Host launcher
// ---------------------------------------------------------------------------
extern "C" void kernel_launch(void* const* d_in, const int* in_sizes, int n_in,
                              void* d_out, int out_size, void* d_ws, size_t ws_size,
                              hipStream_t stream) {
  (void)in_sizes; (void)n_in; (void)out_size; (void)ws_size;
  const float* x  = (const float*)d_in[0];
  const float* wq = (const float*)d_in[1];
  const float* wk = (const float*)d_in[2];
  const float* wv = (const float*)d_in[3];
  const float* wo = (const float*)d_in[4];
  const float* fc = (const float*)d_in[5];
  const float* fs = (const float*)d_in[6];
  // d_in[7] = dense causal mask: unused (masking is computed from indices).
  float* outp = (float*)d_out;

  const int M = Bc * Tc;                          // 4096 rows
  const size_t nX  = (size_t)M * DIMc;            // 8M
  const size_t nWq = (size_t)DIMc * Hc * HDc;     // 4M
  const size_t nWk = (size_t)DIMc * KVHc * HDc;   // 1M
  const size_t nWo = (size_t)(Hc * HDc) * DIMc;   // 4M

  char* p = (char*)d_ws;
  auto alloc = [&](size_t bytes) -> char* {
    char* q = p; p += (bytes + 255) & ~(size_t)255; return q;
  };
  __bf16* xb   = (__bf16*)alloc(nX * 2);
  __bf16* wqb  = (__bf16*)alloc(nWq * 2);
  __bf16* wkb  = (__bf16*)alloc(nWk * 2);
  __bf16* wvb  = (__bf16*)alloc(nWk * 2);
  __bf16* wob  = (__bf16*)alloc(nWo * 2);
  __bf16* q0b  = (__bf16*)alloc((size_t)M * Hc * HDc * 2);
  __bf16* k0b  = (__bf16*)alloc((size_t)M * KVHc * HDc * 2);
  __bf16* v0b  = (__bf16*)alloc((size_t)M * KVHc * HDc * 2);
  __bf16* qTb  = (__bf16*)alloc((size_t)M * Hc * HDc * 2);
  __bf16* kTb  = (__bf16*)alloc((size_t)M * KVHc * HDc * 2);
  __bf16* vTb  = (__bf16*)alloc((size_t)M * KVHc * HDc * 2);
  __bf16* attb = (__bf16*)alloc((size_t)M * Hc * HDc * 2);

  auto cvt = [&](const float* s, __bf16* d, size_t n) {
    cvt_f32_bf16<<<dim3((unsigned)((n + 255) / 256)), dim3(256), 0, stream>>>(s, d, (int)n);
  };
  cvt(x, xb, nX);
  cvt(wq, wqb, nWq);
  cvt(wk, wkb, nWk);
  cvt(wv, wvb, nWk);
  cvt(wo, wob, nWo);

  const unsigned gemmLds = 4u * TILE_B;   // double-buffered A+B tiles

  // QKV projections (bf16 in, bf16 out)
  gemm_wmma<true><<<dim3(2048 / 64, M / 64), 128, gemmLds, stream>>>(xb, wqb, nullptr, q0b, M, 2048, DIMc);
  gemm_wmma<true><<<dim3(512  / 64, M / 64), 128, gemmLds, stream>>>(xb, wkb, nullptr, k0b, M, 512, DIMc);
  gemm_wmma<true><<<dim3(512  / 64, M / 64), 128, gemmLds, stream>>>(xb, wvb, nullptr, v0b, M, 512, DIMc);

  // RoPE + transpose to [b,h,t,hd]; attention scale (1/8) folded into Q.
  {
    int nq = Bc * Tc * Hc * (HDc / 2);
    int nk = Bc * Tc * KVHc * (HDc / 2);
    rope_tr<<<dim3((nq + 255) / 256), 256, 0, stream>>>(q0b, fc, fs, qTb, Hc,   1, 0.125f, nq);
    rope_tr<<<dim3((nk + 255) / 256), 256, 0, stream>>>(k0b, fc, fs, kTb, KVHc, 1, 1.0f,   nk);
    rope_tr<<<dim3((nk + 255) / 256), 256, 0, stream>>>(v0b, fc, fs, vTb, KVHc, 0, 1.0f,   nk);
  }

  // Causal flash attention -> [b,t,h*hd] bf16
  flash_attn<<<dim3(Tc / 64, Bc * Hc), 128, FA_LDS_BYTES, stream>>>(qTb, kTb, vTb, attb);

  // Output projection -> f32 d_out
  gemm_wmma<false><<<dim3(2048 / 64, M / 64), 128, gemmLds, stream>>>(attb, wob, outp, nullptr, M, DIMc, Hc * HDc);
}
